// DeformableDetrTransformerEncoderLayer_18287970746409
// MI455X (gfx1250) — compile-verified, matrix operand order checked by reading, not run on hardware
//
#include <hip/hip_runtime.h>
#include <cstdint>
#include <cstring>

// ---------------- problem constants (hardcoded from reference) ----------------
#define BB 2
#define NN 11253            // 92^2 + 46^2 + 23^2 + 12^2
#define CC 256
#define NHH 8
#define HDD 32
#define NLL 4
#define NPP 4
#define FFF 1024
#define MROWS (BB*NN)       // 22506
#define LN_EPS 1e-5f
#define LDS_PAD 8           // bf16 elements of row padding (16B) to spread LDS banks

typedef __bf16 bf16_t;
typedef bf16_t bf16x16 __attribute__((ext_vector_type(16)));
typedef float  f32x8   __attribute__((ext_vector_type(8)));
typedef uint32_t u32x4 __attribute__((ext_vector_type(4)));
typedef __attribute__((address_space(3))) char lds_char_t;

// ---------------- prep: bf16 casts of query and query+query_pos ----------------
__global__ __launch_bounds__(256)
void prep_kernel(const float* __restrict__ q, const float* __restrict__ qp,
                 bf16_t* __restrict__ qb, bf16_t* __restrict__ qpb, size_t n)
{
    size_t i = (size_t)blockIdx.x * blockDim.x + threadIdx.x;
    if (i < n) {
        float a = q[i];
        qb[i]  = (bf16_t)a;
        qpb[i] = (bf16_t)(a + qp[i]);
    }
}

// ---------------- weight transpose to [N,K] bf16 ----------------
__global__ __launch_bounds__(256)
void wt_kernel(const float* __restrict__ w, bf16_t* __restrict__ wt, int K, int Nc)
{
    size_t t = (size_t)blockIdx.x * blockDim.x + threadIdx.x;
    if (t < (size_t)K * Nc) {
        int k = (int)(t / Nc);
        int n = (int)(t % Nc);
        wt[(size_t)n * K + k] = (bf16_t)w[t];
    }
}

// ---------------- bf16 WMMA GEMM: out[M,Nc] = A[M,K] @ Wt[Nc,K]^T + bias ----------------
// Block = 8 waves covering 32 rows x 128 cols. The 32xK A tile is staged into LDS
// once per block with GLOBAL_LOAD_ASYNC_TO_LDS_B128 (ASYNCcnt), then each wave
// computes two 16x16 M-tiles per B fragment (B-frag loads halved; A-frag loads
// come from LDS with a 16B row pad to avoid 512B-stride bank conflicts).
// Fragment layout per CDNA5 ISA 7.12.2 (16-bit A 16x32):
//   lane<16 : row = lane,    K = {k0..k0+7} U {k0+16..k0+23}
//   lane>=16: row = lane-16, K = {k0+8..k0+15} U {k0+24..k0+31}
// Wt rows are the B-matrix rows (B = W^T stored row-major), loaded identically.
__global__ __launch_bounds__(256)
void gemm_bf16_kernel(const bf16_t* __restrict__ A, const bf16_t* __restrict__ Wt,
                      const float* __restrict__ bias, int M, int K, int Nc,
                      float* __restrict__ outF, bf16_t* __restrict__ outB, int relu)
{
    extern __shared__ char smem_raw[];
    bf16_t* As = (bf16_t*)smem_raw;                 // [32][K + LDS_PAD]
    const int LDSK = K + LDS_PAD;

    const int wave = threadIdx.x >> 5;
    const int lane = threadIdx.x & 31;
    const int r    = lane & 15;
    const int sub  = lane >> 4;          // 0: K {0..7,16..23} ; 1: K {8..15,24..31}
    const int row0 = blockIdx.x * 32;
    const int col0 = blockIdx.y * 128 + wave * 16;

    // ---- async cooperative fill of the 32 x K A tile into LDS ----
    {
        unsigned ldsBase = (unsigned)(uintptr_t)(lds_char_t*)smem_raw;
        const unsigned rowBytes = (unsigned)K * 2u;
        const unsigned totalChunks = (32u * rowBytes) >> 4;   // 16B chunks
        for (unsigned c = threadIdx.x; c < totalChunks; c += 256u) {
            unsigned byteOff  = c << 4;
            unsigned rowLocal = byteOff / rowBytes;
            unsigned inRow    = byteOff - rowLocal * rowBytes;
            int arow = row0 + (int)rowLocal;
            if (arow >= M) arow = M - 1;              // clamp ragged tail (dup rows)
            uint64_t gaddr = (uint64_t)(uintptr_t)((const char*)A
                              + (size_t)arow * rowBytes + inRow);
            unsigned ldsoff = ldsBase + rowLocal * (unsigned)LDSK * 2u + inRow;
            asm volatile("global_load_async_to_lds_b128 %0, %1, off"
                         :: "v"(ldsoff), "v"(gaddr) : "memory");
        }
        asm volatile("s_wait_asynccnt 0x0" ::: "memory");
        __syncthreads();
    }

    const int brow = col0 + r;
    const bf16_t* bptr = Wt + (size_t)brow * K + sub * 8;
    const bf16_t* aptr0 = As + (size_t)(r)      * LDSK + sub * 8;   // M-tile 0
    const bf16_t* aptr1 = As + (size_t)(16 + r) * LDSK + sub * 8;   // M-tile 1

    f32x8 acc0 = {0.f,0.f,0.f,0.f,0.f,0.f,0.f,0.f};
    f32x8 acc1 = {0.f,0.f,0.f,0.f,0.f,0.f,0.f,0.f};

    for (int k0 = 0; k0 < K; k0 += 32) {
        union { bf16x16 v; u32x4 u[2]; } a0, a1, b;
        b.u[0]  = *(const u32x4*)(bptr  + k0);
        b.u[1]  = *(const u32x4*)(bptr  + k0 + 16);
        a0.u[0] = *(const u32x4*)(aptr0 + k0);
        a0.u[1] = *(const u32x4*)(aptr0 + k0 + 16);
        a1.u[0] = *(const u32x4*)(aptr1 + k0);
        a1.u[1] = *(const u32x4*)(aptr1 + k0 + 16);
        acc0 = __builtin_amdgcn_wmma_f32_16x16x32_bf16(
                   false, a0.v, false, b.v, (short)0, acc0, false, false);
        acc1 = __builtin_amdgcn_wmma_f32_16x16x32_bf16(
                   false, a1.v, false, b.v, (short)0, acc1, false, false);
    }

    const int col = col0 + r;
    const float bcol = bias ? bias[col] : 0.f;
    #pragma unroll
    for (int i = 0; i < 8; ++i) {
        // C/D layout: VGPR i -> M = i + 8*sub, N = r
        int rowA = row0 + sub * 8 + i;
        int rowB = row0 + 16 + sub * 8 + i;
        if (rowA < M) {
            float v = acc0[i] + bcol;
            if (outB) {
                if (relu && v < 0.f) v = 0.f;
                outB[(size_t)rowA * Nc + col] = (bf16_t)v;
            } else {
                outF[(size_t)rowA * Nc + col] = v;
            }
        }
        if (rowB < M) {
            float v = acc1[i] + bcol;
            if (outB) {
                if (relu && v < 0.f) v = 0.f;
                outB[(size_t)rowB * Nc + col] = (bf16_t)v;
            } else {
                outF[(size_t)rowB * Nc + col] = v;
            }
        }
    }
}

// ---------------- MSDA: softmax over 16 points + bilinear gather ----------------
// One wave per (b,q,h); lane = channel d (HD == 32 == wave32 width).
__global__ __launch_bounds__(256)
void msda_kernel(const float* __restrict__ value,   // [M, 256] (b,n,h,d)
                 const float* __restrict__ offs,    // [M, 256] (h,l,p,2)
                 const float* __restrict__ attnl,   // [M, 128] (h,l*p)
                 const float* __restrict__ refpts,  // [B,N,NL,2]
                 bf16_t* __restrict__ msda)         // [M, 256] bf16
{
    const int Hs[4]     = {92, 46, 23, 12};
    const int Ws[4]     = {92, 46, 23, 12};
    const int starts[4] = {0, 8464, 10580, 11109};

    int wid  = (int)(((size_t)blockIdx.x * blockDim.x + threadIdx.x) >> 5);
    int lane = threadIdx.x & 31;
    if (wid >= MROWS * NHH) return;
    int mq = wid >> 3;                 // row in [0, M)
    int h  = wid & 7;
    int b  = mq / NN;

    int s = lane & 15;                 // sample index (both wave halves mirror)
    // softmax over the 16 (level,point) logits for this head
    float logit = attnl[(size_t)mq * 128 + h * 16 + s];
    float mx = logit;
    for (int o = 8; o >= 1; o >>= 1) mx = fmaxf(mx, __shfl_xor(mx, o, 32));
    float e = __expf(logit - mx);
    float sm = e;
    for (int o = 8; o >= 1; o >>= 1) sm += __shfl_xor(sm, o, 32);
    float wgt = e / sm;

    // sampling location for sample s (held in lanes s and s+16)
    int   l  = s >> 2;
    float ox = offs[(size_t)mq * 256 + h * 32 + s * 2 + 0];
    float oy = offs[(size_t)mq * 256 + h * 32 + s * 2 + 1];
    float rx = refpts[(size_t)mq * 8 + l * 2 + 0];
    float ry = refpts[(size_t)mq * 8 + l * 2 + 1];
    float lx = rx + ox / (float)Ws[l];
    float ly = ry + oy / (float)Hs[l];

    float acc = 0.f;
    for (int t = 0; t < 16; ++t) {
        float w_t = __shfl(wgt, t, 32);
        float x   = __shfl(lx,  t, 32);
        float y   = __shfl(ly,  t, 32);
        int lt = t >> 2;
        int Wl = Ws[lt], Hl = Hs[lt], st = starts[lt];
        x = x * (float)Wl - 0.5f;
        y = y * (float)Hl - 0.5f;
        float x0f = floorf(x), y0f = floorf(y);
        int x0 = (int)x0f, y0 = (int)y0f;
        float wx = x - x0f, wy = y - y0f;

        float v00 = 0.f, v01 = 0.f, v10 = 0.f, v11 = 0.f;
        #define GATHER(xi, yi, dst)                                              \
            if ((xi) >= 0 && (xi) < Wl && (yi) >= 0 && (yi) < Hl) {              \
                size_t row = (size_t)b * NN + st + (size_t)(yi) * Wl + (xi);     \
                dst = value[row * 256 + h * 32 + lane];                          \
            }
        GATHER(x0,     y0,     v00)
        GATHER(x0 + 1, y0,     v01)
        GATHER(x0,     y0 + 1, v10)
        GATHER(x0 + 1, y0 + 1, v11)
        #undef GATHER

        float top = v00 * (1.f - wx) + v01 * wx;
        float bot = v10 * (1.f - wx) + v11 * wx;
        acc += w_t * (top * (1.f - wy) + bot * wy);
    }
    msda[(size_t)mq * 256 + h * 32 + lane] = (bf16_t)acc;
}

// ---------------- residual add + LayerNorm (block per row, 256 threads) ----------------
__global__ __launch_bounds__(256)
void ln_kernel(const float* __restrict__ a, const float* __restrict__ resid,
               const float* __restrict__ w, const float* __restrict__ bias,
               float* __restrict__ outF, bf16_t* __restrict__ outB)
{
    __shared__ float red[256];
    int row = blockIdx.x;
    int c   = threadIdx.x;
    size_t idx = (size_t)row * CC + c;
    float v = a[idx] + resid[idx];

    red[c] = v; __syncthreads();
    for (int s = 128; s > 0; s >>= 1) { if (c < s) red[c] += red[c + s]; __syncthreads(); }
    float mu = red[0] / (float)CC;
    __syncthreads();

    float d = v - mu;
    red[c] = d * d; __syncthreads();
    for (int s = 128; s > 0; s >>= 1) { if (c < s) red[c] += red[c + s]; __syncthreads(); }
    float var = red[0] / (float)CC;

    float o = d * rsqrtf(var + LN_EPS) * w[c] + bias[c];
    if (outF) outF[idx] = o;
    if (outB) outB[idx] = (bf16_t)o;
}

// ---------------- host-side orchestration ----------------
extern "C" void kernel_launch(void* const* d_in, const int* in_sizes, int n_in,
                              void* d_out, int out_size, void* d_ws, size_t ws_size,
                              hipStream_t stream)
{
    const float* query   = (const float*)d_in[0];
    const float* qpos    = (const float*)d_in[1];
    const float* refpts  = (const float*)d_in[6];
    const float* w_value = (const float*)d_in[7];
    const float* b_value = (const float*)d_in[8];
    const float* w_off   = (const float*)d_in[9];
    const float* b_off   = (const float*)d_in[10];
    const float* w_attn  = (const float*)d_in[11];
    const float* b_attn  = (const float*)d_in[12];
    const float* w_out   = (const float*)d_in[13];
    const float* b_out   = (const float*)d_in[14];
    const float* w_ffn1  = (const float*)d_in[15];
    const float* b_ffn1  = (const float*)d_in[16];
    const float* w_ffn2  = (const float*)d_in[17];
    const float* b_ffn2  = (const float*)d_in[18];
    const float* ln1_w   = (const float*)d_in[19];
    const float* ln1_b   = (const float*)d_in[20];
    const float* ln2_w   = (const float*)d_in[21];
    const float* ln2_b   = (const float*)d_in[22];

    // --- workspace carving (256 B aligned), with buffer reuse once consumed ---
    char* p = (char*)d_ws;
    auto alloc = [&](size_t bytes) -> char* {
        char* r = p; p += (bytes + 255) & ~(size_t)255; return r;
    };
    bf16_t* qb      = (bf16_t*)alloc((size_t)MROWS * CC * 2);
    bf16_t* qpb     = (bf16_t*)alloc((size_t)MROWS * CC * 2);   // reused as x_bf16
    bf16_t* wv_t    = (bf16_t*)alloc((size_t)CC * CC * 2);
    bf16_t* woff_t  = (bf16_t*)alloc((size_t)CC * 256 * 2);
    bf16_t* wattn_t = (bf16_t*)alloc((size_t)CC * 128 * 2);
    bf16_t* wout_t  = (bf16_t*)alloc((size_t)CC * CC * 2);
    bf16_t* wffn1_t = (bf16_t*)alloc((size_t)CC * FFF * 2);
    bf16_t* wffn2_t = (bf16_t*)alloc((size_t)FFF * CC * 2);
    float*  value   = (float*)alloc((size_t)MROWS * CC * 4);    // reused as x_f32
    float*  offs    = (float*)alloc((size_t)MROWS * 256 * 4);   // reused as outproj, then y
    float*  attnl   = (float*)alloc((size_t)MROWS * 128 * 4);
    bf16_t* msda_b  = (bf16_t*)alloc((size_t)MROWS * CC * 2);
    bf16_t* h_b     = (bf16_t*)alloc((size_t)MROWS * FFF * 2);
    (void)ws_size; (void)in_sizes; (void)n_in; (void)out_size;

    float*  x_f32   = value;   // value consumed by msda before LN1 writes here
    float*  outproj = offs;    // offsets consumed by msda before out-proj GEMM
    bf16_t* x_b     = qpb;     // qp consumed by off/attn GEMMs before LN1
    float*  y_f32   = offs;    // outproj consumed by LN1 before FFN2 writes here

    const size_t nElem = (size_t)MROWS * CC;
    const int rowTiles32 = (MROWS + 31) / 32;

    // 1. bf16 activations
    prep_kernel<<<(int)((nElem + 255) / 256), 256, 0, stream>>>(query, qpos, qb, qpb, nElem);

    // 2. weight transposes to [N,K] bf16
    auto wt = [&](const float* w, bf16_t* wtp, int K, int Nc) {
        size_t n = (size_t)K * Nc;
        wt_kernel<<<(int)((n + 255) / 256), 256, 0, stream>>>(w, wtp, K, Nc);
    };
    wt(w_value, wv_t,   CC, CC);
    wt(w_off,   woff_t, CC, 256);
    wt(w_attn,  wattn_t,CC, 128);
    wt(w_out,   wout_t, CC, CC);
    wt(w_ffn1,  wffn1_t,CC, FFF);
    wt(w_ffn2,  wffn2_t,FFF, CC);

    // 3. WMMA GEMMs (dynamic LDS = 32 x (K + pad) bf16 A tile)
    auto gemm = [&](const bf16_t* A, const bf16_t* Wt, const float* bias,
                    int K, int Nc, float* oF, bf16_t* oB, int relu) {
        dim3 grid(rowTiles32, Nc / 128);
        size_t ldsBytes = (size_t)32 * (K + LDS_PAD) * 2;
        gemm_bf16_kernel<<<grid, 256, ldsBytes, stream>>>(A, Wt, bias, MROWS, K, Nc, oF, oB, relu);
    };
    gemm(qb,  wv_t,    b_value, CC, CC,  value, nullptr, 0);   // value projection
    gemm(qpb, woff_t,  b_off,   CC, 256, offs,  nullptr, 0);   // sampling offsets
    gemm(qpb, wattn_t, b_attn,  CC, 128, attnl, nullptr, 0);   // attention logits

    // 4. softmax + deformable bilinear sampling (wave per (b,q,h))
    msda_kernel<<<MROWS, 256, 0, stream>>>(value, offs, attnl, refpts, msda_b);

    // 5. output projection
    gemm(msda_b, wout_t, b_out, CC, CC, outproj, nullptr, 0);

    // 6. residual + LN1 -> x (f32 + bf16)
    ln_kernel<<<MROWS, 256, 0, stream>>>(outproj, query, ln1_w, ln1_b, x_f32, x_b);

    // 7. FFN1 with fused ReLU + bf16 store
    gemm(x_b, wffn1_t, b_ffn1, CC, FFF, nullptr, h_b, 1);

    // 8. FFN2
    gemm(h_b, wffn2_t, b_ffn2, FFF, CC, y_f32, nullptr, 0);

    // 9. residual + LN2 -> d_out (f32)
    ln_kernel<<<MROWS, 256, 0, stream>>>(y_f32, x_f32, ln2_w, ln2_b, (float*)d_out, nullptr);
}
